// OverlapAggregator_42451456754041
// MI455X (gfx1250) — matrix operand order repaired and checked by reading, not compiled, not run on hardware
//
#include <hip/hip_runtime.h>
#include <math.h>

// ---------------------------------------------------------------------------
// OverlapAggregator for MI455X (gfx1250).
// bf16 WMMA (v_wmma_f32_16x16x32_bf16) for all three GEMMs, fp32 everywhere
// else. See analysis: bf16 matrix path (~20us) balances 23.3 TB/s HBM stream
// (~15us); fp32 WMMA (16x16x4) would be ~8x slower on the 53 GFLOP total.
// ---------------------------------------------------------------------------

#define D   768
#define TT  1024
#define SS  8
#define BBATCH 4
#define FF  16
#define KG  1568                 // 2D+F = 1552 padded to multiple of 32
#define MT  (BBATCH*TT)          // 4096 target rows
#define MS  (BBATCH*SS*TT)       // 32768 support rows
#define GW  4                    // waves per GEMM workgroup (128 threads)

typedef __attribute__((ext_vector_type(16))) __bf16 v16bf;
typedef __attribute__((ext_vector_type(8)))  __bf16 v8bf;
typedef __attribute__((ext_vector_type(8)))  float  v8f;
typedef unsigned short u16;
typedef unsigned int   u32;

static __device__ __forceinline__ u16 f2bf(float f) {
    u32 u = __builtin_bit_cast(u32, f);
    u32 r = u + 0x7FFFu + ((u >> 16) & 1u);   // round to nearest even
    return (u16)(r >> 16);
}

static __device__ __forceinline__ float gelu_exact(float x) {
    return 0.5f * x * (1.0f + erff(x * 0.70710678118654752f));
}

// ---- WMMA fragment loaders (wave32, contiguous k order shared by A and B) --------------------
// A (16x32 bf16): lane half h = lane>>4, row m = lane&15.
//   element e -> k = 16*h + e  => one contiguous 16-elem run (matches B).
static __device__ __forceinline__ v16bf load_a_frag(const u16* As, int lda,
                                                    int kk, int lane) {
    const int half = lane >> 4, mr = lane & 15;
    const u16* ap = As + mr * lda + kk + 16 * half;
    v8bf lo = *(const v8bf*)ap;          // k = kk + 16h + 0..7
    v8bf hi = *(const v8bf*)(ap + 8);    // k = kk + 16h + 8..15
    v16bf a;
#pragma unroll
    for (int i = 0; i < 8; ++i) { a[i] = lo[i]; a[i + 8] = hi[i]; }
    return a;
}
// B (32x16 bf16): lane holds column n = lane&15; lanes 0-15: k = kk+0..15,
// lanes 16-31: k = kk+16..31. With W stored transposed (N x K) this is one
// contiguous 32-byte read per lane.
static __device__ __forceinline__ v16bf load_b_frag(const u16* WT, int Kp,
                                                    int n_global, int kk, int lane) {
    const int half = lane >> 4;
    return *(const v16bf*)(WT + (size_t)n_global * Kp + kk + 16 * half);
}

// ---- LayerNorm: one block per row, D = 768 = 256*3 -------------------------
__global__ __launch_bounds__(256)
void ln_kernel(const float* __restrict__ x, const float* __restrict__ g,
               const float* __restrict__ bta, float* __restrict__ out_f,
               u16* __restrict__ out_bf, float* __restrict__ stats) {
    const int row = blockIdx.x, tid = threadIdx.x;
    const float* xr = x + (size_t)row * D;
    float v0 = xr[tid], v1 = xr[tid + 256], v2 = xr[tid + 512];
    __shared__ float ssum[256], ssq[256];
    ssum[tid] = v0 + v1 + v2;
    ssq[tid]  = v0 * v0 + v1 * v1 + v2 * v2;
    __syncthreads();
    for (int off = 128; off > 0; off >>= 1) {
        if (tid < off) { ssum[tid] += ssum[tid + off]; ssq[tid] += ssq[tid + off]; }
        __syncthreads();
    }
    const float mean = ssum[0] * (1.0f / D);
    const float var  = ssq[0] * (1.0f / D) - mean * mean;
    const float rstd = rsqrtf(var + 1e-5f);
    if (stats && tid == 0) { stats[2 * row] = mean; stats[2 * row + 1] = rstd; }
    const float xv[3] = { v0, v1, v2 };
#pragma unroll
    for (int i = 0; i < 3; ++i) {
        const int d = tid + i * 256;
        const float y = (xv[i] - mean) * rstd * g[d] + bta[d];
        if (out_f)  out_f[(size_t)row * D + d] = y;
        if (out_bf) out_bf[(size_t)row * D + d] = f2bf(y);
    }
}

// ---- transpose + convert fp32 weights -> bf16 Wt (N x Kp), zero-pad K ------
__global__ void tcvt_kernel(const float* __restrict__ src, u16* __restrict__ dstT,
                            int K, int Kp, int N, int row_off, int total) {
    for (int i = blockIdx.x * blockDim.x + threadIdx.x; i < total;
         i += gridDim.x * blockDim.x) {
        const int n = i / Kp, k = i % Kp;
        dstT[(size_t)n * Kp + k] =
            (k < K) ? f2bf(src[(size_t)(row_off + k) * N + n]) : (u16)0;
    }
}

// ---- t_score = th_bf16 @ w1a (plain GEMM, fp32 out) ------------------------
__global__ __launch_bounds__(32 * GW)
void gemm_tscore_kernel(const u16* __restrict__ A, const u16* __restrict__ WT,
                        float* __restrict__ Cf) {
    __shared__ u16 As[16 * (D + 8)];
    const int lda = D + 8;
    const int tid = threadIdx.x, lane = tid & 31, wave = tid >> 5;
    const int m0 = blockIdx.x * 16;
    const int vpr = D / 8;
    for (int i = tid; i < 16 * vpr; i += 32 * GW) {
        const int r = i / vpr, c8 = (i % vpr) * 8;
        *(uint4*)(As + r * lda + c8) = *(const uint4*)(A + (size_t)(m0 + r) * D + c8);
    }
    __syncthreads();
    const int half = lane >> 4, nl = lane & 15;
    for (int nt = wave; nt < D / 16; nt += GW) {
        const int ng = nt * 16 + nl;
        v8f c = {};
        for (int kk = 0; kk < D; kk += 32) {
            v16bf a = load_a_frag(As, lda, kk, lane);
            v16bf b = load_b_frag(WT, D, ng, kk, lane);
            c = __builtin_amdgcn_wmma_f32_16x16x32_bf16(false, a, false, b,
                                                        (short)0, c, false, false);
        }
#pragma unroll
        for (int r = 0; r < 8; ++r)
            Cf[(size_t)(m0 + r + 8 * half) * D + ng] = c[r];
    }
}

// ---- fused: h = gelu(sh@w1b + t_score + b1); scores = h . w2 + b2 ----------
__global__ __launch_bounds__(32 * GW)
void score_kernel(const u16* __restrict__ A, const u16* __restrict__ WT,
                  const float* __restrict__ tsc, const float* __restrict__ b1,
                  const float* __restrict__ w2, const float* __restrict__ b2,
                  float* __restrict__ scores) {
    __shared__ u16 As[16 * (D + 8)];
    __shared__ float ssc[16];
    const int lda = D + 8;
    const int tid = threadIdx.x, lane = tid & 31, wave = tid >> 5;
    const int m0 = blockIdx.x * 16;
    const int vpr = D / 8;
    for (int i = tid; i < 16 * vpr; i += 32 * GW) {
        const int r = i / vpr, c8 = (i % vpr) * 8;
        *(uint4*)(As + r * lda + c8) = *(const uint4*)(A + (size_t)(m0 + r) * D + c8);
    }
    if (tid < 16) ssc[tid] = 0.f;
    __syncthreads();
    const int half = lane >> 4, nl = lane & 15;
    float sacc[8] = {0, 0, 0, 0, 0, 0, 0, 0};
    for (int nt = wave; nt < D / 16; nt += GW) {
        const int ng = nt * 16 + nl;
        v8f c = {};
        for (int kk = 0; kk < D; kk += 32) {
            v16bf a = load_a_frag(As, lda, kk, lane);
            v16bf b = load_b_frag(WT, D, ng, kk, lane);
            c = __builtin_amdgcn_wmma_f32_16x16x32_bf16(false, a, false, b,
                                                        (short)0, c, false, false);
        }
        const float w2n = w2[ng], b1n = b1[ng];
#pragma unroll
        for (int r = 0; r < 8; ++r) {
            const int row = m0 + r + 8 * half;          // (b,s,t) row
            const int bb = row >> 13, t = row & (TT - 1);
            const float v = c[r] + tsc[(size_t)((bb << 10) + t) * D + ng] + b1n;
            sacc[r] += gelu_exact(v) * w2n;
        }
    }
#pragma unroll
    for (int r = 0; r < 8; ++r)
        for (int off = 1; off < 16; off <<= 1)
            sacc[r] += __shfl_xor(sacc[r], off, 32);
    if (nl == 0) {
#pragma unroll
        for (int r = 0; r < 8; ++r) atomicAdd(&ssc[r + 8 * half], sacc[r]);
    }
    __syncthreads();
    if (tid < 16) scores[m0 + tid] = ssc[tid] + b2[0];
}

// ---- masked softmax over S, renormalize ------------------------------------
__global__ void softmax_kernel(const float* __restrict__ scores,
                               const unsigned char* __restrict__ mask,
                               float* __restrict__ wsm) {
    const int i = blockIdx.x * blockDim.x + threadIdx.x;
    if (i >= MT) return;
    const int bb = i >> 10, t = i & (TT - 1);
    float sc[SS], mk[SS], mx = -1e30f;
#pragma unroll
    for (int s = 0; s < SS; ++s) {
        const size_t idx = ((size_t)(bb * SS + s) << 10) + t;
        mk[s] = mask[idx] ? 1.f : 0.f;
        sc[s] = mk[s] != 0.f ? scores[idx] : -10000.f;
        mx = fmaxf(mx, sc[s]);
    }
    float sum = 0.f;
#pragma unroll
    for (int s = 0; s < SS; ++s) { sc[s] = expf(sc[s] - mx); sum += sc[s]; }
    const float den = 1.f / sum;
    float sum2 = 0.f;
#pragma unroll
    for (int s = 0; s < SS; ++s) { sc[s] = sc[s] * den * mk[s]; sum2 += sc[s]; }
    const float den2 = 1.f / fmaxf(sum2, 1e-8f);
#pragma unroll
    for (int s = 0; s < SS; ++s)
        wsm[((size_t)(bb * SS + s) << 10) + t] = sc[s] * den2;
}

// ---- summary = sum_s w * LN(support); also build bf16 gate-concat rows -----
__global__ __launch_bounds__(192)
void summary_kernel(const float* __restrict__ xsup, const float* __restrict__ sstats,
                    const float* __restrict__ wsm, const float* __restrict__ sg,
                    const float* __restrict__ sb, const float* __restrict__ th,
                    const float* __restrict__ feat, float* __restrict__ summary,
                    u16* __restrict__ gcat) {
    const int row = blockIdx.x;                 // b*T + t
    const int bb = row >> 10, t = row & (TT - 1);
    const int tid = threadIdx.x, d0 = tid * 4;
    const float4 g4 = *(const float4*)(sg + d0);
    const float4 b4 = *(const float4*)(sb + d0);
    float4 acc = {0.f, 0.f, 0.f, 0.f};
    for (int s = 0; s < SS; ++s) {
        const size_t srow = ((size_t)(bb * SS + s) << 10) + t;
        const float wv = wsm[srow];
        const float mean = sstats[2 * srow], rstd = sstats[2 * srow + 1];
        const float4 x = *(const float4*)(xsup + srow * D + d0);
        acc.x += wv * ((x.x - mean) * rstd * g4.x + b4.x);
        acc.y += wv * ((x.y - mean) * rstd * g4.y + b4.y);
        acc.z += wv * ((x.z - mean) * rstd * g4.z + b4.z);
        acc.w += wv * ((x.w - mean) * rstd * g4.w + b4.w);
    }
    *(float4*)(summary + (size_t)row * D + d0) = acc;
    u16* gr = gcat + (size_t)row * KG;
    const float4 thv = *(const float4*)(th + (size_t)row * D + d0);
    gr[d0]     = f2bf(thv.x); gr[d0 + 1] = f2bf(thv.y);
    gr[d0 + 2] = f2bf(thv.z); gr[d0 + 3] = f2bf(thv.w);
    gr[D + d0]     = f2bf(acc.x); gr[D + d0 + 1] = f2bf(acc.y);
    gr[D + d0 + 2] = f2bf(acc.z); gr[D + d0 + 3] = f2bf(acc.w);
    if (tid < FF)       gr[2 * D + tid] = f2bf(feat[(size_t)row * FF + tid]);
    else if (tid < 32)  gr[2 * D + tid] = 0;   // K padding 1552..1567
}

// ---- fused gate MLP: sigmoid(gelu(gcat@gw1 + b1) . w2 + b2) ---------------
__global__ __launch_bounds__(32 * GW)
void gate_kernel(const u16* __restrict__ A, const u16* __restrict__ WT,
                 const float* __restrict__ b1, const float* __restrict__ w2,
                 const float* __restrict__ b2, float* __restrict__ gate) {
    __shared__ u16 As[16 * (KG + 8)];
    __shared__ float ssc[16];
    const int lda = KG + 8;
    const int tid = threadIdx.x, lane = tid & 31, wave = tid >> 5;
    const int m0 = blockIdx.x * 16;
    const int vpr = KG / 8;                    // 196
    for (int i = tid; i < 16 * vpr; i += 32 * GW) {
        const int r = i / vpr, c8 = (i % vpr) * 8;
        *(uint4*)(As + r * lda + c8) = *(const uint4*)(A + (size_t)(m0 + r) * KG + c8);
    }
    if (tid < 16) ssc[tid] = 0.f;
    __syncthreads();
    const int half = lane >> 4, nl = lane & 15;
    float sacc[8] = {0, 0, 0, 0, 0, 0, 0, 0};
    for (int nt = wave; nt < D / 16; nt += GW) {
        const int ng = nt * 16 + nl;
        v8f c = {};
        for (int kk = 0; kk < KG; kk += 32) {
            v16bf a = load_a_frag(As, lda, kk, lane);
            v16bf b = load_b_frag(WT, KG, ng, kk, lane);
            c = __builtin_amdgcn_wmma_f32_16x16x32_bf16(false, a, false, b,
                                                        (short)0, c, false, false);
        }
        const float w2n = w2[ng], b1n = b1[ng];
#pragma unroll
        for (int r = 0; r < 8; ++r)
            sacc[r] += gelu_exact(c[r] + b1n) * w2n;
    }
#pragma unroll
    for (int r = 0; r < 8; ++r)
        for (int off = 1; off < 16; off <<= 1)
            sacc[r] += __shfl_xor(sacc[r], off, 32);
    if (nl == 0) {
#pragma unroll
        for (int r = 0; r < 8; ++r) atomicAdd(&ssc[r + 8 * half], sacc[r]);
    }
    __syncthreads();
    if (tid < 16) gate[m0 + tid] = 1.f / (1.f + expf(-(ssc[tid] + b2[0])));
}

// ---- residual mix + final LayerNorm ---------------------------------------
__global__ __launch_bounds__(256)
void final_kernel(const float* __restrict__ th, const float* __restrict__ summary,
                  const float* __restrict__ gate, const float* __restrict__ og,
                  const float* __restrict__ ob, float* __restrict__ out,
                  float* __restrict__ out_gate) {
    const int row = blockIdx.x, tid = threadIdx.x;
    const float gv = gate[row];
    float y[3], s = 0.f, sq = 0.f;
#pragma unroll
    for (int i = 0; i < 3; ++i) {
        const int d = tid + i * 256;
        const float tv = th[(size_t)row * D + d];
        const float sm = summary[(size_t)row * D + d];
        const float f = tv + (gv * sm + (1.f - gv) * tv);   // th + fused
        y[i] = f; s += f; sq += f * f;
    }
    __shared__ float ssum[256], ssq[256];
    ssum[tid] = s; ssq[tid] = sq;
    __syncthreads();
    for (int off = 128; off > 0; off >>= 1) {
        if (tid < off) { ssum[tid] += ssum[tid + off]; ssq[tid] += ssq[tid + off]; }
        __syncthreads();
    }
    const float mean = ssum[0] * (1.0f / D);
    const float var  = ssq[0] * (1.0f / D) - mean * mean;
    const float rstd = rsqrtf(var + 1e-5f);
#pragma unroll
    for (int i = 0; i < 3; ++i) {
        const int d = tid + i * 256;
        out[(size_t)row * D + d] = (y[i] - mean) * rstd * og[d] + ob[d];
    }
    if (tid == 0) out_gate[row] = gv;
}

// ---------------------------------------------------------------------------
extern "C" void kernel_launch(void* const* d_in, const int* in_sizes, int n_in,
                              void* d_out, int out_size, void* d_ws, size_t ws_size,
                              hipStream_t stream) {
    const float* target  = (const float*)d_in[0];
    const float* support = (const float*)d_in[1];
    const unsigned char* mask = (const unsigned char*)d_in[2];   // jnp bool: 1B
    const float* feat    = (const float*)d_in[3];
    const float* t_g = (const float*)d_in[4],  *t_b = (const float*)d_in[5];
    const float* s_g = (const float*)d_in[6],  *s_b = (const float*)d_in[7];
    const float* o_g = (const float*)d_in[8],  *o_b = (const float*)d_in[9];
    const float* score_w1 = (const float*)d_in[10], *score_b1 = (const float*)d_in[11];
    const float* score_w2 = (const float*)d_in[12], *score_b2 = (const float*)d_in[13];
    const float* gate_w1  = (const float*)d_in[14], *gate_b1  = (const float*)d_in[15];
    const float* gate_w2  = (const float*)d_in[16], *gate_b2  = (const float*)d_in[17];

    // workspace carve-out (~113 MB total), 256B aligned slices
    char* base = (char*)d_ws;
    size_t off = 0;
    auto alloc = [&](size_t bytes) -> void* {
        void* p = base + off; off += (bytes + 255) & ~(size_t)255; return p;
    };
    float* th      = (float*)alloc((size_t)MT * D * 4);
    u16*   th_bf   = (u16*)  alloc((size_t)MT * D * 2);
    u16*   sh_bf   = (u16*)  alloc((size_t)MS * D * 2);
    float* sstats  = (float*)alloc((size_t)MS * 2 * 4);
    u16*   w1aT    = (u16*)  alloc((size_t)D * D * 2);
    u16*   w1bT    = (u16*)  alloc((size_t)D * D * 2);
    u16*   gw1T    = (u16*)  alloc((size_t)D * KG * 2);
    float* tsc     = (float*)alloc((size_t)MT * D * 4);
    float* scores  = (float*)alloc((size_t)MS * 4);
    float* wsm     = (float*)alloc((size_t)MS * 4);
    float* summary = (float*)alloc((size_t)MT * D * 4);
    u16*   gcat    = (u16*)  alloc((size_t)MT * KG * 2);
    float* gate    = (float*)alloc((size_t)MT * 4);

    // 1-2: LayerNorms (fp32 + bf16 shadow; support keeps mean/rstd stats)
    ln_kernel<<<MT, 256, 0, stream>>>(target, t_g, t_b, th, th_bf, nullptr);
    ln_kernel<<<MS, 256, 0, stream>>>(support, s_g, s_b, nullptr, sh_bf, sstats);

    // 3: weight transpose-convert to bf16 (Wt is N x Kp)
    {
        int tot = D * D;
        tcvt_kernel<<<(tot + 255) / 256, 256, 0, stream>>>(score_w1, w1aT, D, D, D, 0, tot);
        tcvt_kernel<<<(tot + 255) / 256, 256, 0, stream>>>(score_w1, w1bT, D, D, D, D, tot);
        int totg = D * KG;
        tcvt_kernel<<<(totg + 255) / 256, 256, 0, stream>>>(gate_w1, gw1T, 2 * D + FF, KG, D, 0, totg);
    }

    // 4: t_score = th @ w1a
    gemm_tscore_kernel<<<MT / 16, 32 * GW, 0, stream>>>(th_bf, w1aT, tsc);

    // 5: fused support-score GEMM + GELU + dot(w2)
    score_kernel<<<MS / 16, 32 * GW, 0, stream>>>(sh_bf, w1bT, tsc, score_b1,
                                                  score_w2, score_b2, scores);

    // 6: masked softmax over S + renorm
    softmax_kernel<<<(MT + 255) / 256, 256, 0, stream>>>(scores, mask, wsm);

    // 7: weighted summary (re-applies LN via stats) + build gate-concat bf16
    summary_kernel<<<MT, 192, 0, stream>>>(support, sstats, wsm, s_g, s_b,
                                           th, feat, summary, gcat);

    // 8: fused gate MLP -> sigmoid gate
    gate_kernel<<<MT / 16, 32 * GW, 0, stream>>>(gcat, gw1T, gate_b1,
                                                 gate_w2, gate_b2, gate);

    // 9: residual mix + output LN; gate tail after (B,T,D) floats
    final_kernel<<<MT, 256, 0, stream>>>(th, summary, gate, o_g, o_b,
                                         (float*)d_out,
                                         (float*)d_out + (size_t)MT * D);
}
